// RoPEAttention_11330123727239
// MI455X (gfx1250) — matrix operand order repaired and checked by reading, not compiled, not run on hardware
//
#include <hip/hip_runtime.h>

#define DEVI __device__ __forceinline__

typedef __attribute__((ext_vector_type(16))) __bf16 v16bf;
typedef __attribute__((ext_vector_type(8)))  float  v8f;

constexpr int BATCH = 2;
constexpr int T     = 2048;
constexpr int D     = 1024;
constexpr int H     = 16;
constexpr int DH    = 64;
constexpr int ROWS  = BATCH * T;   // 4096

// ---------------------------------------------------------------------------
// CDNA5 async copy: global -> LDS, 16B per lane, tracked by ASYNCcnt
// ---------------------------------------------------------------------------
DEVI void cp16_async(void* lds_ptr, const void* gptr) {
  // Low 32 bits of a generic LDS address are the wave-relative LDS offset.
  unsigned lds_off = (unsigned)(size_t)lds_ptr;
  unsigned long long ga = (unsigned long long)(size_t)gptr;
  asm volatile("global_load_async_to_lds_b128 %0, %1, off"
               :: "v"(lds_off), "v"(ga) : "memory");
}
DEVI void async_wait0() { asm volatile("s_wait_asynccnt 0" ::: "memory"); }
DEVI void ds_wait0()    { asm volatile("s_wait_dscnt 0" ::: "memory"); }

// ---------------------------------------------------------------------------
// WMMA helpers (CDNA5 16x16x32 bf16, f32 accumulate)
// ---------------------------------------------------------------------------
DEVI v8f wmma_bf16(v16bf a, v16bf b, v8f c) {
  return __builtin_amdgcn_wmma_f32_16x16x32_bf16(
      false, a, false, b, (short)0, c, false, false);
}

// A-matrix (16x32 bf16) per-lane K base for pair v: half0 K{0..15},
// half1 K{16..31} (matches B). Two contiguous 16B
// chunks per lane -> compiler merges into 2x ds_load_b128.
DEVI int a_kbase(int v, int half) {
  return (half << 4) + (v << 1);
}

DEVI v16bf lda_bf16(const __bf16* __restrict__ row, int half) {
  v16bf a;
#pragma unroll
  for (int v = 0; v < 8; ++v) {
    int k = a_kbase(v, half);
    a[2 * v]     = row[k];
    a[2 * v + 1] = row[k + 1];
  }
  return a;
}

// B tile: lane = column (lane&15), K = (lane>>4)*16 + e, K contiguous at p.
DEVI v16bf ldb_bf16(const __bf16* __restrict__ p) {
  union { uint4 u[2]; v16bf v; } t;
  t.u[0] = *(const uint4*)p;
  t.u[1] = *(const uint4*)(p + 8);
  return t.v;
}

// ---------------------------------------------------------------------------
// Converters: x -> bf16 row-major; W (f32 [k][n]) -> bf16 W^T [n][k]
// ---------------------------------------------------------------------------
__global__ __launch_bounds__(256) void k_xconv(const float* __restrict__ x,
                                               __bf16* __restrict__ xb) {
  size_t i = (size_t)blockIdx.x * 256 + threadIdx.x;  // one packed pair each
  float2 f = *(const float2*)(x + 2 * i);
  union { __bf16 h[2]; unsigned u; } p;
  p.h[0] = (__bf16)f.x;
  p.h[1] = (__bf16)f.y;
  ((unsigned*)xb)[i] = p.u;
}

__global__ __launch_bounds__(256) void k_wtrans(
    const float* __restrict__ wq, const float* __restrict__ wk,
    const float* __restrict__ wv, const float* __restrict__ wo,
    __bf16* __restrict__ qT, __bf16* __restrict__ kT,
    __bf16* __restrict__ vT, __bf16* __restrict__ oT) {
  __shared__ __bf16 tile[32][33];
  const int m = blockIdx.z;
  const float* src = (m == 0) ? wq : (m == 1) ? wk : (m == 2) ? wv : wo;
  __bf16* dst = (m == 0) ? qT : (m == 1) ? kT : (m == 2) ? vT : oT;
  const int tx = threadIdx.x, ty = threadIdx.y;   // (32,8)
  const int k0 = blockIdx.y * 32, n0 = blockIdx.x * 32;
#pragma unroll
  for (int i = 0; i < 4; ++i)
    tile[ty + i * 8][tx] = (__bf16)src[(size_t)(k0 + ty + i * 8) * D + n0 + tx];
  __syncthreads();
#pragma unroll
  for (int i = 0; i < 4; ++i)
    dst[(size_t)(n0 + ty + i * 8) * D + k0 + tx] = tile[tx][ty + i * 8];
}

// ---------------------------------------------------------------------------
// Staging: NR rows x 32 bf16 (64B/row) -> NR*4 16B chunks over 128 threads
// ---------------------------------------------------------------------------
template <int NR>
DEVI void stage_rows32(__bf16* dst, const __bf16* src, size_t stride, int tid) {
#pragma unroll
  for (int i = 0; i < NR / 32; ++i) {
    int c   = tid * (NR / 32) + i;
    int r   = c >> 2;
    int seg = (c & 3) * 8;
    cp16_async(dst + r * 32 + seg, src + (size_t)r * stride + seg);
  }
}

// ---------------------------------------------------------------------------
// GEMM step: block tile 128(M) x 64(N), 4 waves; wave computes 32x64.
// All operand ds-loads grouped ahead of an 8-WMMA burst.
// ---------------------------------------------------------------------------
DEVI void gemm_step(const __bf16* As, const __bf16* Bs,
                    int wid, int half, int ln, v8f acc[2][4]) {
  v16bf a0 = lda_bf16(As + (wid * 32 + ln) * 32, half);
  v16bf a1 = lda_bf16(As + (wid * 32 + 16 + ln) * 32, half);
  v16bf b0 = ldb_bf16(Bs + (0 * 16 + ln) * 32 + half * 16);
  v16bf b1 = ldb_bf16(Bs + (1 * 16 + ln) * 32 + half * 16);
  v16bf b2 = ldb_bf16(Bs + (2 * 16 + ln) * 32 + half * 16);
  v16bf b3 = ldb_bf16(Bs + (3 * 16 + ln) * 32 + half * 16);
  acc[0][0] = wmma_bf16(a0, b0, acc[0][0]);
  acc[0][1] = wmma_bf16(a0, b1, acc[0][1]);
  acc[0][2] = wmma_bf16(a0, b2, acc[0][2]);
  acc[0][3] = wmma_bf16(a0, b3, acc[0][3]);
  acc[1][0] = wmma_bf16(a1, b0, acc[1][0]);
  acc[1][1] = wmma_bf16(a1, b1, acc[1][1]);
  acc[1][2] = wmma_bf16(a1, b2, acc[1][2]);
  acc[1][3] = wmma_bf16(a1, b3, acc[1][3]);
}

// ---------------------------------------------------------------------------
// Kernel 1: q/k/v projections (+RoPE on q,k). blockIdx.z: 0=q,1=k,2=v.
// ---------------------------------------------------------------------------
__global__ __launch_bounds__(128) void k_proj_rope(
    const __bf16* __restrict__ xb, const __bf16* __restrict__ wqT,
    const __bf16* __restrict__ wkT, const __bf16* __restrict__ wvT,
    __bf16* __restrict__ qb, __bf16* __restrict__ kb, __bf16* __restrict__ vtb) {
  __shared__ __bf16 As[2][128 * 32];   // 16 KB
  __shared__ __bf16 Bs[2][64 * 32];    //  8 KB

  const int mode = blockIdx.z;
  const __bf16* wT = (mode == 0) ? wqT : (mode == 1) ? wkT : wvT;

  const int tid  = threadIdx.x;
  const int lane = tid & 31;
  const int wid  = tid >> 5;
  const int half = lane >> 4;
  const int ln   = lane & 15;

  const int row0 = blockIdx.y * 128;
  const int col0 = blockIdx.x * 64;

  v8f acc[2][4];
#pragma unroll
  for (int mt = 0; mt < 2; ++mt)
#pragma unroll
    for (int nt = 0; nt < 4; ++nt)
#pragma unroll
      for (int r = 0; r < 8; ++r) acc[mt][nt][r] = 0.f;

  stage_rows32<128>(As[0], xb + (size_t)row0 * D, D, tid);
  stage_rows32<64>(Bs[0], wT + (size_t)col0 * D, D, tid);
  async_wait0();
  __syncthreads();

  for (int k0 = 0; k0 < D; k0 += 32) {
    const int buf = (k0 >> 5) & 1;
    if (k0 + 32 < D) {
      stage_rows32<128>(As[buf ^ 1], xb + (size_t)row0 * D + k0 + 32, D, tid);
      stage_rows32<64>(Bs[buf ^ 1], wT + (size_t)col0 * D + k0 + 32, D, tid);
    }
    gemm_step(As[buf], Bs[buf], wid, half, ln, acc);
    async_wait0();
    __syncthreads();
  }

#pragma unroll
  for (int mt = 0; mt < 2; ++mt) {
    const int rbase = row0 + wid * 32 + mt * 16;
    const int bidx  = rbase / T;
    const int trow0 = rbase % T;
#pragma unroll
    for (int nt = 0; nt < 4; ++nt) {
      const int j  = col0 + nt * 16 + ln;  // feature in [0,1024)
      const int hh = j >> 6;
      const int dd = j & 63;
      const float sgn = (j & 1) ? 1.f : -1.f;
      const float invf = __expf(-(float)(j >> 1) * (2.f * 9.210340372f / (float)D));
#pragma unroll
      for (int r = 0; r < 8; ++r) {
        const int t = trow0 + r + 8 * half;
        float val = acc[mt][nt][r];
        float o;
        if (mode < 2) {
          float partner = __shfl_xor(val, 1, 32);  // x[j^1], same row
          float fr = (float)t * invf;
          float c = __cosf(fr), s = __sinf(fr);
          o = val * c + sgn * partner * s;
        } else {
          o = val;
        }
        if (mode == 0)
          qb[(((size_t)bidx * H + hh) * T + t) * DH + dd] = (__bf16)o;
        else if (mode == 1)
          kb[(((size_t)bidx * H + hh) * T + t) * DH + dd] = (__bf16)o;
        else
          vtb[(((size_t)bidx * H + hh) * DH + dd) * T + t] = (__bf16)o;
      }
    }
  }
}

// ---------------------------------------------------------------------------
// Kernel 2: streaming-softmax causal attention, K/V async-staged in LDS.
// grid (T/64, B*H), 128 threads. All waves run the block-uniform kv count;
// fully-masked tiles are numerically inert (rmax=-1e30 => alpha=1, p=0).
// ---------------------------------------------------------------------------
__global__ __launch_bounds__(128) void k_attn(
    const __bf16* __restrict__ qb, const __bf16* __restrict__ kb,
    const __bf16* __restrict__ vtb, __bf16* __restrict__ attn) {
  __shared__ __bf16 Ks[2][32 * 64];   // [kv][dh]
  __shared__ __bf16 Vs[2][64 * 32];   // [dh][kv]
  __shared__ __bf16 Pl[4][16 * 32];   // per-wave P transpose buffer

  const int tid  = threadIdx.x;
  const int lane = tid & 31;
  const int wid  = tid >> 5;
  const int half = lane >> 4;
  const int ln   = lane & 15;

  const int bh    = blockIdx.y;
  const int q0blk = blockIdx.x * 64;
  const int q0    = q0blk + wid * 16;

  const __bf16* Q  = qb  + (size_t)bh * T * DH;
  const __bf16* K  = kb  + (size_t)bh * T * DH;
  const __bf16* Vt = vtb + (size_t)bh * DH * T;

  v16bf aq0 = lda_bf16(Q + (size_t)(q0 + ln) * DH, half);
  v16bf aq1 = lda_bf16(Q + (size_t)(q0 + ln) * DH + 32, half);

  float mrun[8], lrun[8];
  v8f o[4];
#pragma unroll
  for (int r = 0; r < 8; ++r) { mrun[r] = -1e30f; lrun[r] = 0.f; }
#pragma unroll
  for (int nt = 0; nt < 4; ++nt)
#pragma unroll
    for (int r = 0; r < 8; ++r) o[nt][r] = 0.f;

  const int nkt = ((q0blk + 63) >> 5) + 1;  // block-uniform causal trip count

  // prologue: kv tile 0 (K tile is fully contiguous: 32 rows x 64 dh)
#pragma unroll
  for (int i = 0; i < 2; ++i) {
    int c = tid * 2 + i;
    cp16_async(Ks[0] + c * 8, K + c * 8);
  }
  stage_rows32<64>(Vs[0], Vt, T, tid);
  async_wait0();
  __syncthreads();

  for (int kt = 0; kt < nkt; ++kt) {
    const int kvb = kt * 32;
    const int buf = kt & 1;
    if (kt + 1 < nkt) {
#pragma unroll
      for (int i = 0; i < 2; ++i) {
        int c = tid * 2 + i;
        cp16_async(Ks[buf ^ 1] + c * 8, K + (size_t)(kvb + 32) * DH + c * 8);
      }
      stage_rows32<64>(Vs[buf ^ 1], Vt + kvb + 32, T, tid);
    }

    // S = Q @ K^T (16x32 tile): grouped loads, then 4-WMMA burst
    const __bf16* kc0 = Ks[buf] + (0 * 16 + ln) * DH + half * 16;
    const __bf16* kc1 = Ks[buf] + (1 * 16 + ln) * DH + half * 16;
    v16bf bk00 = ldb_bf16(kc0);
    v16bf bk01 = ldb_bf16(kc0 + 32);
    v16bf bk10 = ldb_bf16(kc1);
    v16bf bk11 = ldb_bf16(kc1 + 32);
    v8f s[2];
#pragma unroll
    for (int nt = 0; nt < 2; ++nt)
#pragma unroll
      for (int r = 0; r < 8; ++r) s[nt][r] = 0.f;
    s[0] = wmma_bf16(aq0, bk00, s[0]);
    s[0] = wmma_bf16(aq1, bk01, s[0]);
    s[1] = wmma_bf16(aq0, bk10, s[1]);
    s[1] = wmma_bf16(aq1, bk11, s[1]);

    // scale + causal mask + row-max
    const float sc = 0.125f;  // 1/sqrt(64)
    float rmax[8];
#pragma unroll
    for (int r = 0; r < 8; ++r) {
      const int qi = q0 + r + 8 * half;
      float v0 = s[0][r] * sc;
      float v1 = s[1][r] * sc;
      if (kvb + ln > qi)      v0 = -1e30f;
      if (kvb + 16 + ln > qi) v1 = -1e30f;
      s[0][r] = v0; s[1][r] = v1;
      rmax[r] = fmaxf(v0, v1);
    }
#pragma unroll
    for (int off = 1; off < 16; off <<= 1)
#pragma unroll
      for (int r = 0; r < 8; ++r)
        rmax[r] = fmaxf(rmax[r], __shfl_xor(rmax[r], off, 32));

    float alpha[8], rsum[8];
#pragma unroll
    for (int r = 0; r < 8; ++r) {
      float mnew = fmaxf(mrun[r], rmax[r]);
      alpha[r] = __expf(mrun[r] - mnew);
      mrun[r] = mnew;
      float p0 = __expf(s[0][r] - mnew);
      float p1 = __expf(s[1][r] - mnew);
      s[0][r] = p0; s[1][r] = p1;
      rsum[r] = p0 + p1;
    }
#pragma unroll
    for (int off = 1; off < 16; off <<= 1)
#pragma unroll
      for (int r = 0; r < 8; ++r)
        rsum[r] += __shfl_xor(rsum[r], off, 32);
#pragma unroll
    for (int r = 0; r < 8; ++r) lrun[r] = lrun[r] * alpha[r] + rsum[r];
#pragma unroll
    for (int nt = 0; nt < 4; ++nt)
#pragma unroll
      for (int r = 0; r < 8; ++r) o[nt][r] *= alpha[r];

    // transpose P (C layout -> A layout) through per-wave LDS
    __bf16* L = Pl[wid];
#pragma unroll
    for (int nt = 0; nt < 2; ++nt)
#pragma unroll
      for (int r = 0; r < 8; ++r)
        L[(r + 8 * half) * 32 + nt * 16 + ln] = (__bf16)s[nt][r];
    ds_wait0();

    // grouped loads then 4-WMMA burst: O += P @ V
    v16bf pa  = lda_bf16(L + ln * 32, half);
    v16bf bv0 = ldb_bf16(Vs[buf] + (0 * 16 + ln) * 32 + half * 16);
    v16bf bv1 = ldb_bf16(Vs[buf] + (1 * 16 + ln) * 32 + half * 16);
    v16bf bv2 = ldb_bf16(Vs[buf] + (2 * 16 + ln) * 32 + half * 16);
    v16bf bv3 = ldb_bf16(Vs[buf] + (3 * 16 + ln) * 32 + half * 16);
    o[0] = wmma_bf16(pa, bv0, o[0]);
    o[1] = wmma_bf16(pa, bv1, o[1]);
    o[2] = wmma_bf16(pa, bv2, o[2]);
    o[3] = wmma_bf16(pa, bv3, o[3]);

    async_wait0();
    __syncthreads();
  }

  const int bidx = bh >> 4;
  const int hh   = bh & 15;
#pragma unroll
  for (int r = 0; r < 8; ++r) {
    const float invl = 1.f / lrun[r];
    const int t = q0 + r + 8 * half;
#pragma unroll
    for (int nt = 0; nt < 4; ++nt) {
      const int j = hh * 64 + nt * 16 + ln;
      attn[((size_t)bidx * T + t) * D + j] = (__bf16)(o[nt][r] * invl);
    }
  }
}

// ---------------------------------------------------------------------------
// Kernel 3: out = attn(bf16) @ wo^T(bf16) -> f32
// ---------------------------------------------------------------------------
__global__ __launch_bounds__(128) void k_out_proj(
    const __bf16* __restrict__ attn, const __bf16* __restrict__ woT,
    float* __restrict__ out) {
  __shared__ __bf16 As[2][128 * 32];
  __shared__ __bf16 Bs[2][64 * 32];

  const int tid  = threadIdx.x;
  const int lane = tid & 31;
  const int wid  = tid >> 5;
  const int half = lane >> 4;
  const int ln   = lane & 15;

  const int row0 = blockIdx.y * 128;
  const int col0 = blockIdx.x * 64;

  v8f acc[2][4];
#pragma unroll
  for (int mt = 0; mt < 2; ++mt)
#pragma unroll
    for (int nt = 0; nt < 4; ++nt)
#pragma unroll
      for (int r = 0; r < 8; ++r) acc[mt][nt][r] = 0.f;

  stage_rows32<128>(As[0], attn + (size_t)row0 * D, D, tid);
  stage_rows32<64>(Bs[0], woT + (size_t)col0 * D, D, tid);
  async_wait0();
  __syncthreads();

  for (int k0 = 0; k0 < D; k0 += 32) {
    const int buf = (k0 >> 5) & 1;
    if (k0 + 32 < D) {
      stage_rows32<128>(As[buf ^ 1], attn + (size_t)row0 * D + k0 + 32, D, tid);
      stage_rows32<64>(Bs[buf ^ 1], woT + (size_t)col0 * D + k0 + 32, D, tid);
    }
    gemm_step(As[buf], Bs[buf], wid, half, ln, acc);
    async_wait0();
    __syncthreads();
  }

#pragma unroll
  for (int mt = 0; mt < 2; ++mt) {
    const int rbase = row0 + wid * 32 + mt * 16;
#pragma unroll
    for (int r = 0; r < 8; ++r) {
      const int row = rbase + r + 8 * half;
#pragma unroll
      for (int nt = 0; nt < 4; ++nt)
        out[(size_t)row * D + col0 + nt * 16 + ln] = acc[mt][nt][r];
    }
  }
}

// ---------------------------------------------------------------------------
// Host launcher
// ---------------------------------------------------------------------------
extern "C" void kernel_launch(void* const* d_in, const int* in_sizes, int n_in,
                              void* d_out, int out_size, void* d_ws, size_t ws_size,
                              hipStream_t stream) {
  const float* x  = (const float*)d_in[0];
  // d_in[1] = causal mask (implicit in kernel 2)
  const float* wq = (const float*)d_in[2];
  const float* wk = (const float*)d_in[3];
  const float* wv = (const float*)d_in[4];
  const float* wo = (const float*)d_in[5];
  float* out = (float*)d_out;

  char* ws = (char*)d_ws;
  const size_t ACT = (size_t)ROWS * D * sizeof(__bf16);  // 8 MB
  const size_t WMT = (size_t)D * D * sizeof(__bf16);     // 2 MB
  __bf16* xb   = (__bf16*)(ws);
  __bf16* wqT  = (__bf16*)(ws + ACT);
  __bf16* wkT  = (__bf16*)(ws + ACT + WMT);
  __bf16* wvT  = (__bf16*)(ws + ACT + 2 * WMT);
  __bf16* woT  = (__bf16*)(ws + ACT + 3 * WMT);
  __bf16* qb   = (__bf16*)(ws + ACT + 4 * WMT);
  __bf16* kb   = (__bf16*)(ws + 2 * ACT + 4 * WMT);
  __bf16* vtb  = (__bf16*)(ws + 3 * ACT + 4 * WMT);
  __bf16* attn = (__bf16*)(ws + 4 * ACT + 4 * WMT);

  k_xconv<<<dim3((ROWS * D / 2) / 256), dim3(256), 0, stream>>>(x, xb);
  k_wtrans<<<dim3(D / 32, D / 32, 4), dim3(32, 8), 0, stream>>>(
      wq, wk, wv, wo, wqT, wkT, wvT, woT);
  k_proj_rope<<<dim3(D / 64, ROWS / 128, 3), dim3(128), 0, stream>>>(
      xb, wqT, wkT, wvT, qb, kb, vtb);
  k_attn<<<dim3(T / 64, BATCH * H), dim3(128), 0, stream>>>(qb, kb, vtb, attn);
  k_out_proj<<<dim3(D / 64, ROWS / 128), dim3(128), 0, stream>>>(attn, woT, out);
}
